// GGNN_47132971107214
// MI455X (gfx1250) — matrix operand order repaired and checked by reading, not compiled, not run on hardware
//
#include <hip/hip_runtime.h>
#include <hip/hip_bf16.h>
#include <math.h>

// ---------------------------------------------------------------------------
// CDNA5 WMMA helpers (wave32, v_wmma_f32_16x16x32_f16)
// ---------------------------------------------------------------------------
typedef __attribute__((ext_vector_type(16))) _Float16 v16h;
typedef __attribute__((ext_vector_type(8)))  float    v8f;

#define WMMA_F16(a, b, c) \
  __builtin_amdgcn_wmma_f32_16x16x32_f16(false, (a), false, (b), (short)0, (c), false, false)

// Position of logical element (row m, column c) inside a fragment-ordered
// A buffer. Fragment layout per ISA 7.12.2 (16-bit A 16x32):
//   lane = hl*16 + m ; element i ; K = kt*32 + (i<8 ? hl*8+i : 16+hl*8+(i-8))
// Inverse of the K mapping for kp = c & 31:
//   hl = (kp>>3)&1 ; i = (kp&7) | ((kp>>4)<<3)
__device__ __forceinline__ int apos(int m, int c) {
  int kt = c >> 5, kp = c & 31;
  int hl = (kp >> 3) & 1;
  int i  = (kp & 7) | ((kp >> 4) << 3);
  return ((kt * 32) + hl * 16 + m) * 16 + i;
}

// Load 2 M-tiles x 4 K-tiles of A fragments from a packed LDS buffer.
__device__ __forceinline__ void loadA2(const _Float16* buf, int lane, v16h a[2][4]) {
#pragma unroll
  for (int t = 0; t < 2; ++t)
#pragma unroll
    for (int ks = 0; ks < 4; ++ks)
      a[t][ks] = *(const v16h*)(buf + t * 2048 + (ks * 32 + lane) * 16);
}

// ---------------------------------------------------------------------------
// Utility kernels
// ---------------------------------------------------------------------------
__global__ void zero_f32_kernel(float* __restrict__ p, int n) {
  int i = blockIdx.x * blockDim.x + threadIdx.x;
  if (i < n) p[i] = 0.0f;
}

// Repack a row-major f32 weight W[K][N] into f16 WMMA B-fragment order.
// Fragment f = kt*(N/16) + nt holds 512 f16: (lane l, elem e) ->
//   Krow = kt*32 + (e<8 ? (l>>4)*8+e : 16+(l>>4)*8+(e-8)) , Ncol = nt*16 + (l&15).
// Rows >= K (K padded to multiple of 32) are zero.
__global__ void pack_w16_kernel(const float* __restrict__ W, _Float16* __restrict__ P,
                                int K, int N, int total) {
  int i = blockIdx.x * blockDim.x + threadIdx.x;
  if (i < total) {
    int f = i >> 9, l = (i >> 4) & 31, e = i & 15;
    int nn = N >> 4;
    int kt = f / nn, nt = f - kt * nn;
    int krow = kt * 32 + (e < 8 ? (l >> 4) * 8 + e : 16 + (l >> 4) * 8 + (e - 8));
    int ncol = nt * 16 + (l & 15);
    P[i] = (_Float16)((krow < K) ? W[krow * N + ncol] : 0.0f);
  }
}

// ---------------------------------------------------------------------------
// Edge message kernel: one wave per 32-edge double-tile (2 WMMA M-tiles).
//   x  = concat(h_src[src[e]], h_dst[dst[e]])          [32,128]
//   h1 = relu(x @ W1 + b1); h2 = relu(h1 @ W2 + b2); m = h2 @ W3 + b3
//   out[i] = sum_j (feat@Qw + Qb)[i,j] * m[j] -> scatter to out_msg[dst[e]]
// Two M-tiles share each B fragment -> halves L2 weight traffic per WMMA.
// ---------------------------------------------------------------------------
__global__ __launch_bounds__(64) void edge_mlp_kernel(
    const int* __restrict__ src, const int* __restrict__ dst,
    const float* __restrict__ h_src, const float* __restrict__ h_dst,
    const float* __restrict__ feat,
    const float* __restrict__ Qw, const float* __restrict__ Qb,
    const _Float16* __restrict__ W1p, const float* __restrict__ B1,
    const _Float16* __restrict__ W2p, const float* __restrict__ B2,
    const _Float16* __restrict__ W3p, const float* __restrict__ B3,
    float* __restrict__ out_msg, int E, int paired) {
  __shared__ __align__(32) _Float16 xs_s[2][4096];  // 2 M-tiles, packed A order
  __shared__ __align__(32) _Float16 hs_s[2][4096];
  __shared__ float ms_s[2][512];
  __shared__ float qw_s[1024];
  __shared__ float qb_s[256];

  const int wave = threadIdx.x >> 5;
  const int lane = threadIdx.x & 31;
  const int wt = blockIdx.x * 2 + wave;
  const int e0 = wt * 32;
  _Float16* XS = xs_s[wave];
  _Float16* HS = hs_s[wave];
  float*    MS = ms_s[wave];

  __builtin_prefetch(W1p, 0, 3);  // global_prefetch_b8

  // Stage Qw/Qb into LDS (shared by both waves of the block).
  for (int idx = threadIdx.x; idx < 1024; idx += 64) qw_s[idx] = Qw[idx];
  for (int idx = threadIdx.x; idx < 256; idx += 64) qb_s[idx] = Qb[idx];

  // Stage x into LDS directly in A-fragment order (f16).
  for (int idx = lane; idx < 4096; idx += 32) {
    int r = idx >> 7, c = idx & 127;
    int e = e0 + r;
    float v = 0.0f;
    if (e < E)
      v = (c < 64) ? h_src[src[e] * 64 + c] : h_dst[dst[e] * 64 + (c - 64)];
    XS[(r >> 4) * 2048 + apos(r & 15, c)] = (_Float16)v;
  }
  __syncthreads();

  const int hl = lane >> 4, nI = lane & 15;
  v16h a[2][4];

  // Layer 1: XS @ W1 -> HS (relu)
  loadA2(XS, lane, a);
  for (int nt = 0; nt < 8; ++nt) {
    float bv = B1[nt * 16 + nI];
    v8f c0, c1;
#pragma unroll
    for (int r = 0; r < 8; ++r) { c0[r] = bv; c1[r] = bv; }
#pragma unroll
    for (int ks = 0; ks < 4; ++ks) {
      v16h b = *(const v16h*)(W1p + ((ks * 8 + nt) * 32 + lane) * 16);
      c0 = WMMA_F16(a[0][ks], b, c0);
      c1 = WMMA_F16(a[1][ks], b, c1);
    }
#pragma unroll
    for (int r = 0; r < 8; ++r) {
      int p = apos(r + 8 * hl, nt * 16 + nI);
      HS[p]        = (_Float16)fmaxf(c0[r], 0.0f);
      HS[2048 + p] = (_Float16)fmaxf(c1[r], 0.0f);
    }
  }
  __syncthreads();

  // Layer 2: HS @ W2 -> XS (relu, overwrite staging buffer)
  loadA2(HS, lane, a);
  for (int nt = 0; nt < 8; ++nt) {
    float bv = B2[nt * 16 + nI];
    v8f c0, c1;
#pragma unroll
    for (int r = 0; r < 8; ++r) { c0[r] = bv; c1[r] = bv; }
#pragma unroll
    for (int ks = 0; ks < 4; ++ks) {
      v16h b = *(const v16h*)(W2p + ((ks * 8 + nt) * 32 + lane) * 16);
      c0 = WMMA_F16(a[0][ks], b, c0);
      c1 = WMMA_F16(a[1][ks], b, c1);
    }
#pragma unroll
    for (int r = 0; r < 8; ++r) {
      int p = apos(r + 8 * hl, nt * 16 + nI);
      XS[p]        = (_Float16)fmaxf(c0[r], 0.0f);
      XS[2048 + p] = (_Float16)fmaxf(c1[r], 0.0f);
    }
  }
  __syncthreads();

  // Layer 3: XS @ W3 -> MS ([32,16], single N tile per M-tile)
  {
    loadA2(XS, lane, a);
    float bv = B3[nI];
    v8f c0, c1;
#pragma unroll
    for (int r = 0; r < 8; ++r) { c0[r] = bv; c1[r] = bv; }
#pragma unroll
    for (int ks = 0; ks < 4; ++ks) {
      v16h b = *(const v16h*)(W3p + (ks * 32 + lane) * 16);
      c0 = WMMA_F16(a[0][ks], b, c0);
      c1 = WMMA_F16(a[1][ks], b, c1);
    }
#pragma unroll
    for (int r = 0; r < 8; ++r) {
      int m = r + 8 * hl;
      MS[m * 16 + nI]        = c0[r];
      MS[(16 + m) * 16 + nI] = c1[r];
    }
  }
  __syncthreads();

  // Per-edge q-transform + scatter. 512 (edge,i) items over 32 lanes.
  for (int tt = 0; tt < 16; ++tt) {
    int item = lane + tt * 32;
    int r = item >> 4, i = item & 15;  // r = 2*tt + hl
    int e = e0 + r;
    float acc = 0.0f;
    if (e < E) {
      float f0 = feat[e * 4 + 0], f1 = feat[e * 4 + 1];
      float f2 = feat[e * 4 + 2], f3 = feat[e * 4 + 3];
#pragma unroll
      for (int j = 0; j < 16; ++j) {
        int col = i * 16 + j;
        float q = f0 * qw_s[col] + f1 * qw_s[256 + col] + f2 * qw_s[512 + col] +
                  f3 * qw_s[768 + col] + qb_s[col];
        acc += q * MS[r * 16 + j];
      }
    }
    if (paired) {
      // Edges 2f, 2f+1 share a destination factor and sit in lanes L, L^16.
      float partner = __shfl_xor(acc, 16, 32);
      if (hl == 0 && e < E) out_msg[dst[e] * 16 + i] = acc + partner;
    } else if (e < E) {
      atomicAdd(&out_msg[dst[e] * 16 + i], acc);
    }
  }
}

// ---------------------------------------------------------------------------
// GRU kernel (gate order r,z,n), 16 nodes per wave, fully register-resident:
//   gi = msg @ Wih + bih (K=16 zero-padded to 32), gh = h @ Whh + bhh (K=64)
// All 12 gate tiles stay in VGPRs; gate math reuses the D-fragment layout:
// tiles nt, nt+4, nt+8 hold r/z/n entries for the same (node,d) position.
// ---------------------------------------------------------------------------
__global__ __launch_bounds__(64) void gru_kernel(
    const float* __restrict__ msg,      // [N,16]
    float* __restrict__ h,              // [N,64] in place
    const _Float16* __restrict__ Wihp,  // packed [pad32,192] -> 12 frags
    const _Float16* __restrict__ Whhp,  // packed [64,192]    -> 24 frags
    const float* __restrict__ bih, const float* __restrict__ bhh, int N) {
  __shared__ __align__(32) _Float16 am_s[2][512];
  __shared__ __align__(32) _Float16 ah_s[2][1024];

  const int wave = threadIdx.x >> 5;
  const int lane = threadIdx.x & 31;
  const int tile = blockIdx.x * 2 + wave;
  const int n0 = tile * 16;
  _Float16* AM = am_s[wave];
  _Float16* AH = ah_s[wave];

  for (int idx = lane; idx < 512; idx += 32) {
    int r = idx >> 5, c = idx & 31;
    int node = n0 + r;
    float v = (node < N && c < 16) ? msg[node * 16 + c] : 0.0f;
    AM[apos(r, c)] = (_Float16)v;
  }
  for (int idx = lane; idx < 1024; idx += 32) {
    int r = idx >> 6, c = idx & 63;
    int node = n0 + r;
    float v = (node < N) ? h[node * 64 + c] : 0.0f;
    AH[apos(r, c)] = (_Float16)v;
  }
  __syncthreads();

  const int hl = lane >> 4, nI = lane & 15;

  v16h am = *(const v16h*)(AM + lane * 16);
  v16h ah[2];
#pragma unroll
  for (int ks = 0; ks < 2; ++ks) ah[ks] = *(const v16h*)(AH + (ks * 32 + lane) * 16);

  v8f grz[8], gin[4], ghn[4];
#pragma unroll
  for (int nt = 0; nt < 12; ++nt) {
    int ncol = nt * 16 + nI;
    v8f cg, ch;
    float bg = bih[ncol], bh = bhh[ncol];
#pragma unroll
    for (int r = 0; r < 8; ++r) { cg[r] = bg; ch[r] = bh; }
    {
      v16h b = *(const v16h*)(Wihp + (nt * 32 + lane) * 16);
      cg = WMMA_F16(am, b, cg);
    }
#pragma unroll
    for (int ks = 0; ks < 2; ++ks) {
      v16h b = *(const v16h*)(Whhp + ((ks * 12 + nt) * 32 + lane) * 16);
      ch = WMMA_F16(ah[ks], b, ch);
    }
    if (nt < 8) {
#pragma unroll
      for (int r = 0; r < 8; ++r) grz[nt][r] = cg[r] + ch[r];
    } else {
      gin[nt - 8] = cg;
      ghn[nt - 8] = ch;
    }
  }

  // Gate math in registers: (lane nI, row r, half hl) <-> node n0+r+8*hl,
  // dim d = dt*16+nI ; tiles dt / dt+4 / dt+8 give r / z / n gate entries.
#pragma unroll
  for (int dt = 0; dt < 4; ++dt) {
    int d = dt * 16 + nI;
#pragma unroll
    for (int r = 0; r < 8; ++r) {
      int node = n0 + r + 8 * hl;
      if (node < N) {
        float rg = 1.0f / (1.0f + expf(-grz[dt][r]));
        float zg = 1.0f / (1.0f + expf(-grz[dt + 4][r]));
        float ng = tanhf(gin[dt][r] + rg * ghn[dt][r]);
        float ho = h[node * 64 + d];
        h[node * 64 + d] = (1.0f - zg) * ng + zg * ho;
      }
    }
  }
}

// ---------------------------------------------------------------------------
// Readout: relu(relu(h @ W1 + b1) @ W2 + b2) @ W3 + b3 -> softmax over 2 cols
// ---------------------------------------------------------------------------
__global__ __launch_bounds__(64) void readout_kernel(
    const float* __restrict__ h,                                    // [N,64]
    const _Float16* __restrict__ W1p, const float* __restrict__ B1, // [64,128]
    const _Float16* __restrict__ W2p, const float* __restrict__ B2, // [128,128]
    const float* __restrict__ W3, const float* __restrict__ B3,     // [128,2] f32
    float* __restrict__ out, int N) {
  __shared__ __align__(32) _Float16 a0_s[2][1024];
  __shared__ __align__(32) _Float16 h1_s[2][2048];
  __shared__ _Float16 h2_s[2][16 * 128];

  const int wave = threadIdx.x >> 5;
  const int lane = threadIdx.x & 31;
  const int tile = blockIdx.x * 2 + wave;
  const int n0 = tile * 16;
  _Float16* A0 = a0_s[wave];
  _Float16* H1 = h1_s[wave];
  _Float16* H2 = h2_s[wave];

  for (int idx = lane; idx < 1024; idx += 32) {
    int r = idx >> 6, c = idx & 63;
    int node = n0 + r;
    A0[apos(r, c)] = (_Float16)((node < N) ? h[node * 64 + c] : 0.0f);
  }
  __syncthreads();

  const int hl = lane >> 4, nI = lane & 15;

  {  // layer 1, K=64 -> H1 (packed A order for layer 2)
    v16h a[2];
#pragma unroll
    for (int ks = 0; ks < 2; ++ks) a[ks] = *(const v16h*)(A0 + (ks * 32 + lane) * 16);
    for (int nt = 0; nt < 8; ++nt) {
      float bv = B1[nt * 16 + nI];
      v8f c;
#pragma unroll
      for (int r = 0; r < 8; ++r) c[r] = bv;
#pragma unroll
      for (int ks = 0; ks < 2; ++ks) {
        v16h b = *(const v16h*)(W1p + ((ks * 8 + nt) * 32 + lane) * 16);
        c = WMMA_F16(a[ks], b, c);
      }
#pragma unroll
      for (int r = 0; r < 8; ++r)
        H1[apos(r + 8 * hl, nt * 16 + nI)] = (_Float16)fmaxf(c[r], 0.0f);
    }
  }
  __syncthreads();

  {  // layer 2, K=128 -> H2 (row-major, consumed by VALU layer 3)
    v16h a[4];
#pragma unroll
    for (int ks = 0; ks < 4; ++ks) a[ks] = *(const v16h*)(H1 + (ks * 32 + lane) * 16);
    for (int nt = 0; nt < 8; ++nt) {
      float bv = B2[nt * 16 + nI];
      v8f c;
#pragma unroll
      for (int r = 0; r < 8; ++r) c[r] = bv;
#pragma unroll
      for (int ks = 0; ks < 4; ++ks) {
        v16h b = *(const v16h*)(W2p + ((ks * 8 + nt) * 32 + lane) * 16);
        c = WMMA_F16(a[ks], b, c);
      }
#pragma unroll
      for (int r = 0; r < 8; ++r) {
        int m = r + 8 * hl;
        H2[m * 128 + nt * 16 + nI] = (_Float16)fmaxf(c[r], 0.0f);
      }
    }
  }
  __syncthreads();

  if (lane < 16) {  // layer 3 (N=2) + softmax on VALU
    int node = n0 + lane;
    if (node < N) {
      float o0 = B3[0], o1 = B3[1];
      for (int d = 0; d < 128; ++d) {
        float v = (float)H2[lane * 128 + d];
        o0 += v * W3[d * 2 + 0];
        o1 += v * W3[d * 2 + 1];
      }
      float mx = fmaxf(o0, o1);
      float e0 = expf(o0 - mx), e1 = expf(o1 - mx);
      float s = e0 + e1;
      out[node * 2 + 0] = e0 / s;
      out[node * 2 + 1] = e1 / s;
    }
  }
}

// ---------------------------------------------------------------------------
// Host-side orchestration
// ---------------------------------------------------------------------------
static inline int ceil_div(int a, int b) { return (a + b - 1) / b; }

extern "C" void kernel_launch(void* const* d_in, const int* in_sizes, int n_in,
                              void* d_out, int out_size, void* d_ws, size_t ws_size,
                              hipStream_t stream) {
  const int*   v2f_src  = (const int*)d_in[0];
  const int*   v2f_dst  = (const int*)d_in[1];
  const int*   f2v_src  = (const int*)d_in[2];
  const int*   f2v_dst  = (const int*)d_in[3];
  const float* feat_v2f = (const float*)d_in[4];
  const float* feat_f2v = (const float*)d_in[5];
  // d_in[6], d_in[7]: n_nodes / n_factors device scalars (derived on host below)
  const float* Qw = (const float*)d_in[8];
  const float* Qb = (const float*)d_in[9];
  const float* mp1_w1 = (const float*)d_in[10]; const float* mp1_b1 = (const float*)d_in[11];
  const float* mp1_w2 = (const float*)d_in[12]; const float* mp1_b2 = (const float*)d_in[13];
  const float* mp1_w3 = (const float*)d_in[14]; const float* mp1_b3 = (const float*)d_in[15];
  const float* mp2_w1 = (const float*)d_in[16]; const float* mp2_b1 = (const float*)d_in[17];
  const float* mp2_w2 = (const float*)d_in[18]; const float* mp2_b2 = (const float*)d_in[19];
  const float* mp2_w3 = (const float*)d_in[20]; const float* mp2_b3 = (const float*)d_in[21];
  const float* g1_wih = (const float*)d_in[22]; const float* g1_whh = (const float*)d_in[23];
  const float* g1_bih = (const float*)d_in[24]; const float* g1_bhh = (const float*)d_in[25];
  const float* g2_wih = (const float*)d_in[26]; const float* g2_whh = (const float*)d_in[27];
  const float* g2_bih = (const float*)d_in[28]; const float* g2_bhh = (const float*)d_in[29];
  const float* ro_w1 = (const float*)d_in[30]; const float* ro_b1 = (const float*)d_in[31];
  const float* ro_w2 = (const float*)d_in[32]; const float* ro_b2 = (const float*)d_in[33];
  const float* ro_w3 = (const float*)d_in[34]; const float* ro_b3 = (const float*)d_in[35];

  const int E  = in_sizes[0];   // 2 * n_factors
  const int NN = out_size / 2;  // n_nodes
  const int NF = E / 2;         // n_factors

  // Workspace carve-up (256B aligned).
  char* ws = (char*)d_ws;
  size_t off = 0;
  auto alloc = [&](size_t bytes) -> void* {
    void* p = ws + off;
    off = (off + bytes + 255) & ~(size_t)255;
    return p;
  };
  float* fac_h   = (float*)alloc((size_t)NF * 64 * 4);
  float* var_h   = (float*)alloc((size_t)NN * 64 * 4);
  float* fac_msg = (float*)alloc((size_t)NF * 16 * 4);
  float* var_msg = (float*)alloc((size_t)NN * 16 * 4);
  // Packed f16 weights: (padK/32)*(N/16) fragments of 512 f16.
  _Float16* w1a  = (_Float16*)alloc(32 * 512 * 2);  // mp1_w1 128x128
  _Float16* w2a  = (_Float16*)alloc(32 * 512 * 2);  // mp1_w2 128x128
  _Float16* w3a  = (_Float16*)alloc(4 * 512 * 2);   // mp1_w3 128x16
  _Float16* w1b  = (_Float16*)alloc(32 * 512 * 2);
  _Float16* w2b  = (_Float16*)alloc(32 * 512 * 2);
  _Float16* w3b  = (_Float16*)alloc(4 * 512 * 2);
  _Float16* wih1 = (_Float16*)alloc(12 * 512 * 2);  // 16(->32)x192
  _Float16* whh1 = (_Float16*)alloc(24 * 512 * 2);  // 64x192
  _Float16* wih2 = (_Float16*)alloc(12 * 512 * 2);
  _Float16* whh2 = (_Float16*)alloc(24 * 512 * 2);
  _Float16* rw1  = (_Float16*)alloc(16 * 512 * 2);  // 64x128
  _Float16* rw2  = (_Float16*)alloc(32 * 512 * 2);  // 128x128
  (void)ws_size; (void)n_in;

  auto pack = [&](const float* s, _Float16* d, int K, int N) {
    int total = ((K + 31) / 32) * (N / 16) * 512;
    pack_w16_kernel<<<ceil_div(total, 256), 256, 0, stream>>>(s, d, K, N, total);
  };
  pack(mp1_w1, w1a, 128, 128); pack(mp1_w2, w2a, 128, 128); pack(mp1_w3, w3a, 128, 16);
  pack(mp2_w1, w1b, 128, 128); pack(mp2_w2, w2b, 128, 128); pack(mp2_w3, w3b, 128, 16);
  pack(g1_wih, wih1, 16, 192); pack(g1_whh, whh1, 64, 192);
  pack(g2_wih, wih2, 16, 192); pack(g2_whh, whh2, 64, 192);
  pack(ro_w1, rw1, 64, 128);   pack(ro_w2, rw2, 128, 128);

  zero_f32_kernel<<<ceil_div(NF * 64, 256), 256, 0, stream>>>(fac_h, NF * 64);
  zero_f32_kernel<<<ceil_div(NN * 64, 256), 256, 0, stream>>>(var_h, NN * 64);

  const int eblocks = ceil_div(ceil_div(E, 32), 2);   // 32 edges per wave
  const int fblocks = ceil_div(ceil_div(NF, 16), 2);
  const int vblocks = ceil_div(ceil_div(NN, 16), 2);

  for (int step = 0; step < 10; ++step) {
    // var -> factor (paired deterministic scatter fully overwrites fac_msg)
    edge_mlp_kernel<<<eblocks, 64, 0, stream>>>(
        v2f_src, v2f_dst, var_h, fac_h, feat_v2f, Qw, Qb,
        w1a, mp1_b1, w2a, mp1_b2, w3a, mp1_b3, fac_msg, E, /*paired=*/1);
    gru_kernel<<<fblocks, 64, 0, stream>>>(fac_msg, fac_h, wih1, whh1, g1_bih, g1_bhh, NF);
    // factor -> var (atomic scatter; needs zeroed buffer)
    zero_f32_kernel<<<ceil_div(NN * 16, 256), 256, 0, stream>>>(var_msg, NN * 16);
    edge_mlp_kernel<<<eblocks, 64, 0, stream>>>(
        f2v_src, f2v_dst, fac_h, var_h, feat_f2v, Qw, Qb,
        w1b, mp2_b1, w2b, mp2_b2, w3b, mp2_b3, var_msg, E, /*paired=*/0);
    gru_kernel<<<vblocks, 64, 0, stream>>>(var_msg, var_h, wih2, whh2, g2_bih, g2_bhh, NN);
  }

  readout_kernel<<<vblocks, 64, 0, stream>>>(
      var_h, rw1, ro_b1, rw2, ro_b2, ro_w3, ro_b3, (float*)d_out, NN);
}
